// BasicBlock_28312424415645
// MI455X (gfx1250) — 16-bit and 8-bit matrix operand order repaired and checked by reading, compiled, not run on hardware
//
#include <hip/hip_runtime.h>

typedef _Float16 h8   __attribute__((ext_vector_type(8)));
typedef _Float16 v16h __attribute__((ext_vector_type(16)));
typedef float    v8f  __attribute__((ext_vector_type(8)));
typedef int      v8i  __attribute__((ext_vector_type(8)));
typedef int      v4i  __attribute__((ext_vector_type(4)));

#define CIN   128
#define COUT  128
#define HWDIM 28
#define PIX   784
#define PHW   30
#define PPIX  900
#define BATCH 64
#define MTOT  (BATCH*PIX)
#define PTOT  (BATCH*PPIX)
#define KSZ   (CIN*9)
#define LDSH  1160
#define LDSB  1168
#define BN_EPS 1e-5f

__global__ void k_init(unsigned* s) {
    if (threadIdx.x < 2) s[threadIdx.x] = 0u;
}

__global__ void k_padx(const float* __restrict__ x, _Float16* __restrict__ xp) {
    int e = blockIdx.x * blockDim.x + threadIdx.x;
    int c = e & 127, p = e >> 7;
    int b = p / PPIX, r = p - b * PPIX;
    int ph = r / PHW, pw = r - ph * PHW;
    float v = 0.f;
    if (ph >= 1 && ph <= HWDIM && pw >= 1 && pw <= HWDIM)
        v = x[(size_t)(b * CIN + c) * PIX + (ph - 1) * HWDIM + (pw - 1)];
    xp[e] = (_Float16)v;
}

__global__ void k_maxabs(const float* __restrict__ w, const float* __restrict__ mask,
                         int n, unsigned* __restrict__ out) {
    __shared__ float red[256];
    float mx = 0.f;
    for (int i = blockIdx.x * blockDim.x + threadIdx.x; i < n; i += gridDim.x * blockDim.x) {
        float v = w[i];
        if (mask) v = (mask[i] > 0.5f) ? v : 0.f;
        mx = fmaxf(mx, fabsf(v));
    }
    red[threadIdx.x] = mx; __syncthreads();
    for (int s = 128; s > 0; s >>= 1) {
        if (threadIdx.x < s) red[threadIdx.x] = fmaxf(red[threadIdx.x], red[threadIdx.x + s]);
        __syncthreads();
    }
    if (threadIdx.x == 0) atomicMax(out, __float_as_uint(red[0]));
}

__global__ void k_qw_f16(const float* __restrict__ w, const unsigned* __restrict__ mx,
                         _Float16* __restrict__ out) {
    int e = blockIdx.x * blockDim.x + threadIdx.x;
    int n = e / KSZ, r = e - n * KSZ, pos = r >> 7, c = r & 127;
    float tm = tanhf(__uint_as_float(*mx));
    float t  = tanhf(w[(size_t)(n * CIN + c) * 9 + pos]);
    float q  = rintf(15.f * (t / (2.f * tm) + 0.5f));
    out[e] = (_Float16)(2.f * q * (1.f / 15.f) - 1.f);
}

__global__ void k_qw_i8(const float* __restrict__ w, const float* __restrict__ mask,
                        const unsigned* __restrict__ mx, signed char* __restrict__ out) {
    int e = blockIdx.x * blockDim.x + threadIdx.x;
    int n = e / KSZ, r = e - n * KSZ, pos = r >> 7, c = r & 127;
    size_t src = (size_t)(n * CIN + c) * 9 + pos;
    float v = (mask[src] > 0.5f) ? w[src] : 0.f;
    float tm = tanhf(__uint_as_float(*mx));
    float t  = tanhf(v);
    int qi = (int)rintf(15.f * (t / (2.f * tm) + 0.5f));
    out[e] = (signed char)(2 * qi - 15);
}

__global__ __launch_bounds__(256)
void k_conv1(const _Float16* __restrict__ xp, const _Float16* __restrict__ wq,
             float* __restrict__ out) {
    __shared__ _Float16 bsh[16 * LDSH];
    int ntile = blockIdx.y;

    for (int i = threadIdx.x; i < 16 * 144; i += 256) {
        int col = i / 144, seg = i - col * 144;
        *(h8*)(bsh + col * LDSH + seg * 8) =
            *(const h8*)(wq + (size_t)(ntile * 16 + col) * KSZ + seg * 8);
    }
    __syncthreads();

    int lane = threadIdx.x & 31;
    int wid  = threadIdx.x >> 5;
    int mtile = blockIdx.x * 8 + wid;
    int l16  = lane & 15;
    bool lo  = lane < 16;

    int mrow = mtile * 16 + l16;
    int b = mrow / PIX, rem = mrow - b * PIX;
    int oh = rem / HWDIM, ow = rem - oh * HWDIM;

    const _Float16* abase = xp + (size_t)(b * PPIX + oh * PHW + ow) * CIN;
    const _Float16* lbase = bsh + l16 * LDSH + (lo ? 0 : 8);
    int aoff = lo ? 0 : 8;
    v8f acc = {};

    for (int kh = 0; kh < 3; ++kh) {
        for (int kw = 0; kw < 3; ++kw) {
            const _Float16* ap = abase + (kh * PHW + kw) * CIN;
            const _Float16* bp = lbase + (kh * 3 + kw) * CIN;
            #pragma unroll
            for (int kc = 0; kc < CIN; kc += 32) {
                union { v16h v; h8 h[2]; } af, bf;
                af.h[0] = *(const h8*)(ap + kc + aoff);
                af.h[1] = *(const h8*)(ap + kc + 16 + aoff);
                bf.h[0] = *(const h8*)(bp + kc);
                bf.h[1] = *(const h8*)(bp + kc + 16);
                acc = __builtin_amdgcn_wmma_f32_16x16x32_f16(
                        false, af.v, false, bf.v, (short)0, acc, false, false);
            }
        }
    }
    float* op = out + (size_t)(mtile * 16 + (lo ? 0 : 8)) * COUT + ntile * 16 + l16;
    #pragma unroll
    for (int r = 0; r < 8; ++r) op[(size_t)r * COUT] = acc[r];
}

__global__ void k_bnstats(const float* __restrict__ src, float scale,
                          float* __restrict__ mean_o, float* __restrict__ rsig_o) {
    int c = blockIdx.x;
    __shared__ float s1[256], s2[256];
    float sum = 0.f, ss = 0.f;
    for (int m = threadIdx.x; m < MTOT; m += 256) {
        float v = src[(size_t)m * COUT + c] * scale;
        sum += v; ss += v * v;
    }
    s1[threadIdx.x] = sum; s2[threadIdx.x] = ss; __syncthreads();
    for (int s = 128; s > 0; s >>= 1) {
        if (threadIdx.x < s) {
            s1[threadIdx.x] += s1[threadIdx.x + s];
            s2[threadIdx.x] += s2[threadIdx.x + s];
        }
        __syncthreads();
    }
    if (threadIdx.x == 0) {
        float mean = s1[0] / (float)MTOT;
        float var  = s2[0] / (float)MTOT - mean * mean;
        mean_o[c] = mean;
        rsig_o[c] = rsqrtf(var + BN_EPS);
    }
}

__global__ void k_act1(const float* __restrict__ src, const float* __restrict__ mean,
                       const float* __restrict__ rsig, const float* __restrict__ gamma,
                       const float* __restrict__ beta, unsigned char* __restrict__ out) {
    int e = blockIdx.x * blockDim.x + threadIdx.x;
    int c = e & 127, p = e >> 7;
    int b = p / PPIX, r = p - b * PPIX;
    int ph = r / PHW, pw = r - ph * PHW;
    unsigned char code = 0;
    if (ph >= 1 && ph <= HWDIM && pw >= 1 && pw <= HWDIM) {
        int m = b * PIX + (ph - 1) * HWDIM + (pw - 1);
        float y = (src[(size_t)m * COUT + c] - mean[c]) * rsig[c] * gamma[c] + beta[c];
        y = fminf(fmaxf(y, 0.f), 1.f);
        code = (unsigned char)(int)rintf(15.f * y);
    }
    out[e] = code;
}

__global__ __launch_bounds__(256)
void k_conv2(const unsigned char* __restrict__ actp, const signed char* __restrict__ wq,
             float* __restrict__ out) {
    __shared__ signed char bsh[16 * LDSB];
    int ntile = blockIdx.y;

    for (int i = threadIdx.x; i < 16 * 72; i += 256) {
        int col = i / 72, seg = i - col * 72;
        *(v4i*)(bsh + col * LDSB + seg * 16) =
            *(const v4i*)(wq + (size_t)(ntile * 16 + col) * KSZ + seg * 16);
    }
    __syncthreads();

    int lane = threadIdx.x & 31;
    int wid  = threadIdx.x >> 5;
    int mtile = blockIdx.x * 8 + wid;
    int l16  = lane & 15;
    bool lo  = lane < 16;

    int mrow = mtile * 16 + l16;
    int b = mrow / PIX, rem = mrow - b * PIX;
    int oh = rem / HWDIM, ow = rem - oh * HWDIM;

    const unsigned char* abase = actp + (size_t)(b * PPIX + oh * PHW + ow) * CIN;
    const signed char*   lbase = bsh + l16 * LDSB + (lo ? 0 : 16);
    int aoff = lo ? 0 : 16;
    v8i acc = {};

    for (int kh = 0; kh < 3; ++kh) {
        for (int kw = 0; kw < 3; ++kw) {
            const unsigned char* ap = abase + (kh * PHW + kw) * CIN;
            const signed char*   bp = lbase + (kh * 3 + kw) * CIN;
            #pragma unroll
            for (int kc = 0; kc < CIN; kc += 64) {
                union { v8i v; unsigned long long q[4]; } af;
                af.q[0] = *(const unsigned long long*)(ap + kc +      aoff);
                af.q[1] = *(const unsigned long long*)(ap + kc +  8 + aoff);
                af.q[2] = *(const unsigned long long*)(ap + kc + 32 + aoff);
                af.q[3] = *(const unsigned long long*)(ap + kc + 40 + aoff);
                union { v8i v; v4i x[2]; } bf;
                bf.x[0] = *(const v4i*)(bp + kc);
                bf.x[1] = *(const v4i*)(bp + kc + 32);
                acc = __builtin_amdgcn_wmma_i32_16x16x64_iu8(
                        false, af.v, true, bf.v, acc, false, false);
            }
        }
    }
    float* op = out + (size_t)(mtile * 16 + (lo ? 0 : 8)) * COUT + ntile * 16 + l16;
    #pragma unroll
    for (int r = 0; r < 8; ++r) op[(size_t)r * COUT] = (float)acc[r];
}

__global__ void k_final(const float* __restrict__ src, const float* __restrict__ x,
                        const float* __restrict__ mean, const float* __restrict__ rsig,
                        const float* __restrict__ gamma, const float* __restrict__ beta,
                        float scale, float* __restrict__ out) {
    int e = blockIdx.x * blockDim.x + threadIdx.x;
    int m = e >> 7, c = e & 127;
    int b = m / PIX, rem = m - b * PIX;
    float v = src[e] * scale;
    float y = (v - mean[c]) * rsig[c] * gamma[c] + beta[c];
    y += x[(size_t)(b * CIN + c) * PIX + rem];
    y = fminf(fmaxf(y, 0.f), 1.f);
    out[(size_t)(b * COUT + c) * PIX + rem] = rintf(15.f * y) * (1.f / 15.f);
}

extern "C" void kernel_launch(void* const* d_in, const int* in_sizes, int n_in,
                              void* d_out, int out_size, void* d_ws, size_t ws_size,
                              hipStream_t stream) {
    const float* x     = (const float*)d_in[0];
    const float* w1    = (const float*)d_in[1];
    const float* w2    = (const float*)d_in[2];
    const float* mask2 = (const float*)d_in[3];
    const float* g1    = (const float*)d_in[4];
    const float* b1    = (const float*)d_in[5];
    const float* g2    = (const float*)d_in[6];
    const float* b2    = (const float*)d_in[7];
    float* out = (float*)d_out;

    char* ws = (char*)d_ws;
    size_t off = 0;
    auto carve = [&](size_t bytes) {
        void* p = ws + off;
        off = (off + bytes + 255) & ~(size_t)255;
        return p;
    };
    _Float16*      xp    = (_Float16*)     carve((size_t)PTOT * CIN * 2);
    _Float16*      w1q   = (_Float16*)     carve((size_t)COUT * KSZ * 2);
    signed char*   w2q   = (signed char*)  carve((size_t)COUT * KSZ);
    float*         convo = (float*)        carve((size_t)MTOT * COUT * 4);
    unsigned char* actp  = (unsigned char*)carve((size_t)PTOT * CIN);
    float*         stats = (float*)        carve(4 * COUT * 4);
    unsigned*      mxs   = (unsigned*)     carve(256);
    float* mean1 = stats, *rsig1 = stats + COUT, *mean2 = stats + 2 * COUT, *rsig2 = stats + 3 * COUT;

    const int EP = (PTOT * CIN) / 256;
    const int EW = (MTOT * CIN) / 256;
    const int WQ = (COUT * KSZ) / 256;
    dim3 cgrid(MTOT / 16 / 8, COUT / 16);

    k_init  <<<1, 32, 0, stream>>>(mxs);
    k_padx  <<<EP, 256, 0, stream>>>(x, xp);
    k_maxabs<<<64, 256, 0, stream>>>(w1, nullptr, COUT * KSZ, mxs + 0);
    k_maxabs<<<64, 256, 0, stream>>>(w2, mask2,   COUT * KSZ, mxs + 1);
    k_qw_f16<<<WQ, 256, 0, stream>>>(w1, mxs + 0, w1q);
    k_qw_i8 <<<WQ, 256, 0, stream>>>(w2, mask2, mxs + 1, w2q);

    k_conv1  <<<cgrid, 256, 0, stream>>>(xp, w1q, convo);
    k_bnstats<<<COUT, 256, 0, stream>>>(convo, 1.0f, mean1, rsig1);
    k_act1   <<<EP, 256, 0, stream>>>(convo, mean1, rsig1, g1, b1, actp);

    k_conv2  <<<cgrid, 256, 0, stream>>>(actp, w2q, convo);
    k_bnstats<<<COUT, 256, 0, stream>>>(convo, 1.f / 225.f, mean2, rsig2);
    k_final  <<<EW, 256, 0, stream>>>(convo, x, mean2, rsig2, g2, b2, 1.f / 225.f, out);
}
